// SEALDGCNN_53420803228461
// MI455X (gfx1250) — matrix operand order repaired and checked by reading, not compiled, not run on hardware
//
#include <hip/hip_runtime.h>

typedef __attribute__((ext_vector_type(16))) _Float16 v16h;
typedef __attribute__((ext_vector_type(8)))  _Float16 v8h;
typedef __attribute__((ext_vector_type(8)))  float    v8f;

#define B_GRAPHS 2048
#define NPG      100
#define MPAD     112          // 7 M-tiles of 16
#define HID      64
#define EPG      800
#define E_TOT    (B_GRAPHS * EPG)
#define LATD     193
#define KTOP     30

// ---- dynamic LDS layout (bytes), segments 16B-aligned through sel; y1/y3/o1 only 8B-aligned ----
#define OFF_LAT   0                               // float lat[100][193]        77200
#define OFF_H     (OFF_LAT  + NPG*LATD*4)         // float h[100][64]           25600
#define OFF_HWT   (OFF_H    + NPG*HID*4)          // float hw_t[64][112]        28672
#define OFF_H16   (OFF_HWT  + HID*MPAD*4)         // _Float16 h16[112][64]      14336
#define OFF_WT    (OFF_H16  + MPAD*HID*2)         // _Float16 Wt[64][64] (n,k)   8192
#define OFF_EP    (OFF_WT   + HID*HID*2)          // uint16_t epack[800]         1600
#define OFF_DINV  (OFF_EP   + EPG*2)              // float dinv[100]
#define OFF_DEG   (OFF_DINV + NPG*4)              // int   degc[100]
#define OFF_HWS   (OFF_DEG  + NPG*4)              // float hws[100]
#define OFF_ACC   (OFF_HWS  + NPG*4)              // float accs[100]
#define OFF_SEL   (OFF_ACC  + NPG*4)              // int   sel[30]
#define OFF_Y1    (OFF_SEL  + KTOP*4)             // float y1[16*30] (reused W3s[64])
#define OFF_Y3    (OFF_Y1   + 16*KTOP*4)          // float y3[352]
#define OFF_O1    (OFF_Y3   + 352*4)              // float o1[128]
#define SMEM_BYTES (OFF_O1  + 128*4)              // = 161,160 B  (2 WGs / 320KB WGP)

union AFrag { v16h v; v8h h[2]; };

__launch_bounds__(256, 1)
__global__ void seal_dgcnn_fused(const int* __restrict__ z,
                                 const int* __restrict__ edge_index,
                                 const float* __restrict__ z_emb,
                                 const float* __restrict__ W0, const float* __restrict__ b0,
                                 const float* __restrict__ W1, const float* __restrict__ b1,
                                 const float* __restrict__ W2, const float* __restrict__ b2,
                                 const float* __restrict__ W3, const float* __restrict__ b3,
                                 const float* __restrict__ c1w, const float* __restrict__ c1b,
                                 const float* __restrict__ c2w, const float* __restrict__ c2b,
                                 const float* __restrict__ l1w, const float* __restrict__ l1b,
                                 const float* __restrict__ l2w, const float* __restrict__ l2b,
                                 float* __restrict__ out)
{
    extern __shared__ char smem[];
    float*     lat   = (float*)(smem + OFF_LAT);
    float*     h     = (float*)(smem + OFF_H);
    float*     hw_t  = (float*)(smem + OFF_HWT);   // [k][m] transposed, stride MPAD
    _Float16*  h16   = (_Float16*)(smem + OFF_H16); // [m][k], rows 100..111 zero
    _Float16*  Wt    = (_Float16*)(smem + OFF_WT);  // [n][k] transposed
    uint16_t*  epack = (uint16_t*)(smem + OFF_EP);
    float*     dinv  = (float*)(smem + OFF_DINV);
    int*       degc  = (int*)(smem + OFF_DEG);
    float*     hws   = (float*)(smem + OFF_HWS);
    float*     accs  = (float*)(smem + OFF_ACC);
    int*       sel   = (int*)(smem + OFF_SEL);
    float*     y1    = (float*)(smem + OFF_Y1);
    float*     y3    = (float*)(smem + OFF_Y3);
    float*     o1    = (float*)(smem + OFF_O1);

    const int g = blockIdx.x;
    const int t = threadIdx.x;
    const int L = t & 31;              // lane in wave32
    const int wave = t >> 5;           // 8 waves
    const int hl  = L >> 4;            // lane half (0/1)
    const int l15 = L & 15;

    // ---------------- degrees + edge staging ----------------
    for (int i = t; i < NPG; i += 256) degc[i] = 0;
    // zero f16 padding rows (100..111) once; never rewritten
    for (int i = t; i < (MPAD - NPG) * HID; i += 256)
        h16[NPG * HID + i] = (_Float16)0.0f;
    __syncthreads();
    for (int e = t; e < EPG; e += 256) {
        int s = edge_index[g * EPG + e]         - g * NPG;   // local [0,100)
        int d = edge_index[E_TOT + g * EPG + e] - g * NPG;
        epack[e] = (uint16_t)(s | (d << 8));
        atomicAdd(&degc[d], 1);
    }
    __syncthreads();
    for (int n = t; n < NPG; n += 256) dinv[n] = rsqrtf((float)degc[n] + 1.0f);

    // ---------------- embedding gather: h = z_emb[z] (f32 + f16 copies) --------
    for (int i = t; i < NPG * HID; i += 256) {
        int n = i >> 6, k = i & 63;
        float v = z_emb[(long)z[g * NPG + n] * HID + k];
        h[i]   = v;
        h16[i] = (_Float16)v;
    }
    __syncthreads();

    // ---------------- GCN layers 0..2 (64x64 via WMMA) ----------------
    for (int l = 0; l < 3; ++l) {
        const float* Wp = (l == 0) ? W0 : ((l == 1) ? W1 : W2);
        const float* bp = (l == 0) ? b0 : ((l == 1) ? b1 : b2);

        // stage W transposed as f16: Wt[n][k] = W[k][n]
        for (int i = t; i < HID * HID; i += 256) {
            int k = i >> 6, n = i & 63;
            Wt[n * HID + k] = (_Float16)Wp[k * HID + n];
        }
        __syncthreads();

        // hw_t[k][m] = (h @ W)^T : 7 M-tiles x 4 N-tiles, K=64 (2 steps of 32)
        for (int tile = wave; tile < 28; tile += 8) {
            const int mt = tile >> 2, nt = tile & 3;
            const int mrow = mt * 16 + l15;        // A row for this lane
            const int ncol = nt * 16 + l15;        // B column for this lane
            v8f c = {};
            #pragma unroll
            for (int ks = 0; ks < 2; ++ks) {
                // A fragment: two contiguous 8xf16 runs (ISA 16-bit 16x32 layout)
                AFrag a;
                const _Float16* ap = h16 + mrow * HID + ks * 32 + hl * 8;
                a.h[0] = *(const v8h*)(ap);
                a.h[1] = *(const v8h*)(ap + 16);
                // B fragment: two contiguous 8xf16 runs (same K order as A)
                AFrag b;
                const _Float16* bp16 = Wt + ncol * HID + ks * 32 + hl * 8;
                b.h[0] = *(const v8h*)(bp16);
                b.h[1] = *(const v8h*)(bp16 + 16);
                c = __builtin_amdgcn_wmma_f32_16x16x32_f16(false, a.v, false, b.v,
                                                           (short)0, c, false, false);
            }
            // C store transposed: 8 consecutive m per lane -> two float4 stores
            const int mb = mt * 16 + hl * 8;
            float* cp = hw_t + ncol * MPAD + mb;
            *(float4*)(cp)     = make_float4(c[0], c[1], c[2], c[3]);
            *(float4*)(cp + 4) = make_float4(c[4], c[5], c[6], c[7]);
        }
        __syncthreads();

        // accumulator init (self-loop + bias), overwrite h (input fully consumed)
        for (int i = t; i < NPG * HID; i += 256) {
            int n = i >> 6, k = i & 63;
            float di = dinv[n];
            h[i] = hw_t[k * MPAD + n] * di * di + bp[k];
        }
        __syncthreads();

        // edge scatter: h[dst][k] += hw[src][k] * dinv[src]*dinv[dst]
        for (int i = t; i < EPG * HID; i += 256) {
            int e = i >> 6, k = i & 63;
            int pk = epack[e];
            int s = pk & 255, d = pk >> 8;
            atomicAdd(&h[d * HID + k], hw_t[k * MPAD + s] * (dinv[s] * dinv[d]));
        }
        __syncthreads();

        // tanh -> next-layer input (f32 + f16) and latent slice
        for (int i = t; i < NPG * HID; i += 256) {
            int n = i >> 6, k = i & 63;
            float v2 = tanhf(h[i]);
            h[i]   = v2;
            h16[i] = (_Float16)v2;
            lat[n * LATD + l * HID + k] = v2;
        }
        __syncthreads();
    }

    // ---------------- layer 3 (64 -> 1) ----------------
    float* W3s = y1;  // reuse scratch
    for (int i = t; i < HID; i += 256) W3s[i] = W3[i];
    __syncthreads();
    for (int n = t; n < NPG; n += 256) {
        float acc = 0.0f;
        for (int k = 0; k < HID; ++k) acc += h[n * HID + k] * W3s[k];
        hws[n] = acc;
    }
    __syncthreads();
    for (int n = t; n < NPG; n += 256) {
        float di = dinv[n];
        accs[n] = hws[n] * di * di + b3[0];
    }
    __syncthreads();
    for (int e = t; e < EPG; e += 256) {
        int pk = epack[e];
        int s = pk & 255, d = pk >> 8;
        atomicAdd(&accs[d], hws[s] * (dinv[s] * dinv[d]));
    }
    __syncthreads();
    for (int n = t; n < NPG; n += 256) lat[n * LATD + 192] = tanhf(accs[n]);
    __syncthreads();

    // ---------------- sort-pool: top-30 by channel 192, stable desc ----------------
    for (int n = t; n < NPG; n += 256) {
        float kn = lat[n * LATD + 192];
        int rank = 0;
        for (int m2 = 0; m2 < NPG; ++m2) {
            float km = lat[m2 * LATD + 192];
            rank += (km > kn) || ((km == kn) && (m2 < n));
        }
        if (rank < KTOP) sel[rank] = n;
    }
    __syncthreads();

    // ---------------- conv1: 16 filters width 193 stride 193 + relu ----------------
    for (int i = t; i < 16 * KTOP; i += 256) {
        int cc = i / KTOP, j = i % KTOP;
        int node = sel[j];
        float acc = c1b[cc];
        for (int l2 = 0; l2 < LATD; ++l2)
            acc += c1w[cc * LATD + l2] * lat[node * LATD + l2];
        y1[cc * KTOP + j] = fmaxf(acc, 0.0f);
    }
    __syncthreads();

    // ---------------- maxpool(2) fused into conv2 (16->32, width 5) + relu ----------
    for (int i = t; i < 352; i += 256) {
        int c2 = i / 11, j = i % 11;
        float acc = c2b[c2];
        for (int c1 = 0; c1 < 16; ++c1) {
            #pragma unroll
            for (int kk = 0; kk < 5; ++kk) {
                int p = j + kk;  // pooled index 0..14
                float a0 = y1[c1 * KTOP + 2 * p];
                float a1 = y1[c1 * KTOP + 2 * p + 1];
                acc += c2w[(c2 * 16 + c1) * 5 + kk] * fmaxf(a0, a1);
            }
        }
        y3[i] = fmaxf(acc, 0.0f);
    }
    __syncthreads();

    // ---------------- lin1 (352->128) + relu ----------------
    for (int u = t; u < 128; u += 256) {
        float acc = l1b[u];
        for (int i = 0; i < 352; ++i) acc += y3[i] * l1w[i * 128 + u];
        o1[u] = fmaxf(acc, 0.0f);
    }
    __syncthreads();

    // ---------------- lin2 (128->1) ----------------
    if (t == 0) {
        float acc = l2b[0];
        for (int u = 0; u < 128; ++u) acc += o1[u] * l2w[u];
        out[g] = acc;
    }
}

extern "C" void kernel_launch(void* const* d_in, const int* in_sizes, int n_in,
                              void* d_out, int out_size, void* d_ws, size_t ws_size,
                              hipStream_t stream) {
    const int*   z     = (const int*)d_in[0];
    const int*   ei    = (const int*)d_in[1];
    // d_in[2] = batch (equal-size graphs; unused)
    const float* z_emb = (const float*)d_in[3];
    const float* W0 = (const float*)d_in[4],  *b0 = (const float*)d_in[5];
    const float* W1 = (const float*)d_in[6],  *b1 = (const float*)d_in[7];
    const float* W2 = (const float*)d_in[8],  *b2 = (const float*)d_in[9];
    const float* W3 = (const float*)d_in[10], *b3 = (const float*)d_in[11];
    const float* c1w = (const float*)d_in[12], *c1b = (const float*)d_in[13];
    const float* c2w = (const float*)d_in[14], *c2b = (const float*)d_in[15];
    const float* l1w = (const float*)d_in[16], *l1b = (const float*)d_in[17];
    const float* l2w = (const float*)d_in[18], *l2b = (const float*)d_in[19];
    float* out = (float*)d_out;

    // Raise dynamic-LDS cap (CDNA5 WGP supports up to 320 KB per workgroup).
    // Host-side attribute set, not a stream op: safe under graph capture.
    (void)hipFuncSetAttribute((const void*)seal_dgcnn_fused,
                              hipFuncAttributeMaxDynamicSharedMemorySize,
                              (int)SMEM_BYTES);

    seal_dgcnn_fused<<<B_GRAPHS, 256, SMEM_BYTES, stream>>>(
        z, ei, z_emb, W0, b0, W1, b1, W2, b2, W3, b3,
        c1w, c1b, c2w, c2b, l1w, l1b, l2w, l2b, out);
    (void)d_ws; (void)ws_size; (void)in_sizes; (void)n_in; (void)out_size;
}
